// DSIB_26980984553579
// MI455X (gfx1250) — matrix operand order repaired and checked by reading, not compiled, not run on hardware
//
#include <hip/hip_runtime.h>
#include <hip/hip_bf16.h>

typedef __attribute__((ext_vector_type(16))) __bf16 v16bf;
typedef __attribute__((ext_vector_type(8)))  float  v8f;

#define BSZ 512
#define NX  64
#define HID 256

// ---------------------------------------------------------------------------
// Prep 1: Ax[j][h] = X[j] @ W1[0:64, h]           (512x256)
//         Ay[i][h] = Y[i] @ W1[64:128, h] + b1[h] (512x256)
// blockIdx.x in [0,1024): first 512 blocks -> Ax, next 512 -> Ay
// ---------------------------------------------------------------------------
__global__ __launch_bounds__(256) void prep_axay(
    const float* __restrict__ X, const float* __restrict__ Y,
    const float* __restrict__ W1, const float* __restrict__ b1,
    float* __restrict__ Ax, float* __restrict__ Ay)
{
    __shared__ float row[NX];
    const bool isY = blockIdx.x >= BSZ;
    const int  r   = blockIdx.x & (BSZ - 1);
    const int  h   = threadIdx.x;

    if (h < NX) row[h] = (isY ? Y : X)[r * NX + h];
    __syncthreads();

    const float* w = W1 + (isY ? NX * HID : 0);
    float s = isY ? b1[h] : 0.0f;
#pragma unroll 8
    for (int k = 0; k < NX; ++k)
        s = fmaf(row[k], w[k * HID + h], s);

    (isY ? Ay : Ax)[r * HID + h] = s;
}

// ---------------------------------------------------------------------------
// Prep 2: pack W2 (f32 [256,256]) into bf16 WMMA B-operand layout.
// B layout (16-bit, 16x16x32): n = nt*16 + (lane&15); k = ks*32 + (t/8)*16 + hi*8 + (t%8).
// Stored so the main kernel loads one contiguous 32B v16bf per (nt,ks,lane).
// ---------------------------------------------------------------------------
__global__ __launch_bounds__(256) void pack_w2(
    const float* __restrict__ W2, __bf16* __restrict__ W2p)
{
    const int id   = blockIdx.x * 256 + threadIdx.x;   // 0..4095
    const int lane = id & 31;
    const int rem  = id >> 5;          // 0..127
    const int ks   = rem & 7;
    const int nt   = rem >> 3;
    const int hi   = (lane >> 4) & 1;
    const int n    = nt * 16 + (lane & 15);

    __bf16* dst = W2p + (size_t)id * 16;
#pragma unroll
    for (int t = 0; t < 16; ++t) {
        const int k = ks * 32 + (t >> 3) * 16 + hi * 8 + (t & 7);
        dst[t] = (__bf16)W2[k * HID + n];
    }
}

// ---------------------------------------------------------------------------
// A-fragment build: lane ml holds row M of the H1 tile.
// halves t: K = (t/8)*16 + hi*8 + (t%8), K-step base folded into k0.
// ---------------------------------------------------------------------------
__device__ __forceinline__ v16bf build_a(const float* __restrict__ axr,
                                         const float* __restrict__ ayr,
                                         int k0)
{
    const float4 xa0 = *(const float4*)(axr + k0);
    const float4 xa1 = *(const float4*)(axr + k0 + 4);
    const float4 xb0 = *(const float4*)(axr + k0 + 16);
    const float4 xb1 = *(const float4*)(axr + k0 + 20);
    const float4 ya0 = *(const float4*)(ayr + k0);
    const float4 ya1 = *(const float4*)(ayr + k0 + 4);
    const float4 yb0 = *(const float4*)(ayr + k0 + 16);
    const float4 yb1 = *(const float4*)(ayr + k0 + 20);

    v16bf a;
    a[0]  = (__bf16)fmaxf(xa0.x + ya0.x, 0.0f);
    a[1]  = (__bf16)fmaxf(xa0.y + ya0.y, 0.0f);
    a[2]  = (__bf16)fmaxf(xa0.z + ya0.z, 0.0f);
    a[3]  = (__bf16)fmaxf(xa0.w + ya0.w, 0.0f);
    a[4]  = (__bf16)fmaxf(xa1.x + ya1.x, 0.0f);
    a[5]  = (__bf16)fmaxf(xa1.y + ya1.y, 0.0f);
    a[6]  = (__bf16)fmaxf(xa1.z + ya1.z, 0.0f);
    a[7]  = (__bf16)fmaxf(xa1.w + ya1.w, 0.0f);
    a[8]  = (__bf16)fmaxf(xb0.x + yb0.x, 0.0f);
    a[9]  = (__bf16)fmaxf(xb0.y + yb0.y, 0.0f);
    a[10] = (__bf16)fmaxf(xb0.z + yb0.z, 0.0f);
    a[11] = (__bf16)fmaxf(xb0.w + yb0.w, 0.0f);
    a[12] = (__bf16)fmaxf(xb1.x + yb1.x, 0.0f);
    a[13] = (__bf16)fmaxf(xb1.y + yb1.y, 0.0f);
    a[14] = (__bf16)fmaxf(xb1.z + yb1.z, 0.0f);
    a[15] = (__bf16)fmaxf(xb1.w + yb1.w, 0.0f);
    return a;
}

// ---------------------------------------------------------------------------
// Main: per wave, 32 pairs (fixed i = y row, j0..j0+31 = x cols).
// All 16 A fragments (2 M-subtiles x 8 K-steps, 128 VGPRs) are built ONCE
// and stay resident; the N dimension is streamed 2 tiles at a time with
// only 32 accumulator VGPRs live, the epilogue folding each N-tile into the
// persistent score partials. Each B fragment from LDS feeds 2 WMMAs.
// ---------------------------------------------------------------------------
__global__ __launch_bounds__(256) void mlp_scores(
    const float* __restrict__ Ax, const float* __restrict__ Ay,
    const __bf16* __restrict__ W2p, const float* __restrict__ b2,
    const float* __restrict__ W3, const float* __restrict__ b3,
    float* __restrict__ scores)
{
    extern __shared__ __bf16 lds[];                    // 65536 halves = 128 KB

    // Stage swizzled W2 into LDS (uint4 = 16B chunks, 8192 total).
    {
        const uint4* src = (const uint4*)W2p;
        uint4*       dst = (uint4*)lds;
        for (int t = threadIdx.x; t < 8192; t += 256) dst[t] = src[t];
    }

    const int lane = threadIdx.x & 31;
    const int wave = threadIdx.x >> 5;
    const int i    = blockIdx.y;                       // y row
    const int j0   = blockIdx.x * 256 + wave * 32;     // x col base
    const int hi   = lane >> 4;
    const int ml   = lane & 15;                        // A-row / C-col index

    const float* ax0 = Ax + (size_t)(j0 + ml) * HID;
    const float* ax1 = Ax + (size_t)(j0 + 16 + ml) * HID;
    const float* ayr = Ay + (size_t)i * HID;

    // Build all A fragments while the LDS staging drains.
    v16bf afrag0[8], afrag1[8];
#pragma unroll
    for (int ks = 0; ks < 8; ++ks) {
        const int k0 = ks * 32 + hi * 8;
        afrag0[ks] = build_a(ax0, ayr, k0);
        afrag1[ks] = build_a(ax1, ayr, k0);
    }
    __syncthreads();

    float p0[8], p1[8];
#pragma unroll
    for (int r = 0; r < 8; ++r) { p0[r] = 0.0f; p1[r] = 0.0f; }

    const __bf16* lbase = lds + (size_t)lane * 16;

#pragma unroll 2
    for (int nt = 0; nt < 16; ++nt) {
        v8f acc0, acc1;
#pragma unroll
        for (int r = 0; r < 8; ++r) { acc0[r] = 0.0f; acc1[r] = 0.0f; }

        const __bf16* lnt = lbase + (size_t)nt * (8 * 32 * 16);
#pragma unroll
        for (int ks = 0; ks < 8; ++ks) {
            const v16bf bm = *(const v16bf*)(lnt + (size_t)ks * (32 * 16));
            acc0 = __builtin_amdgcn_wmma_f32_16x16x32_bf16(
                false, afrag0[ks], false, bm, (short)0, acc0, false, false);
            acc1 = __builtin_amdgcn_wmma_f32_16x16x32_bf16(
                false, afrag1[ks], false, bm, (short)0, acc1, false, false);
        }

        // Fused epilogue for this N-tile: h2 = relu(acc + b2[n]);
        // partial += h2 * W3[n]. C layout: reg r, lane -> (M=r+8*hi, N=ml).
        const int n = nt * 16 + ml;
        const float b2v = b2[n];
        const float w3v = W3[n];
#pragma unroll
        for (int r = 0; r < 8; ++r) {
            p0[r] = fmaf(fmaxf(acc0[r] + b2v, 0.0f), w3v, p0[r]);
            p1[r] = fmaf(fmaxf(acc1[r] + b2v, 0.0f), w3v, p1[r]);
        }
    }

    // Reduce over the 16 N-lanes of each half-wave (xor of bits 0..3 stays
    // inside the group).
#pragma unroll
    for (int r = 0; r < 8; ++r) {
        float a = p0[r], b = p1[r];
        a += __shfl_xor(a, 1, 32);  b += __shfl_xor(b, 1, 32);
        a += __shfl_xor(a, 2, 32);  b += __shfl_xor(b, 2, 32);
        a += __shfl_xor(a, 4, 32);  b += __shfl_xor(b, 4, 32);
        a += __shfl_xor(a, 8, 32);  b += __shfl_xor(b, 8, 32);
        p0[r] = a; p1[r] = b;
    }
    if (ml == 0) {
        const float bb3 = b3[0];
        float* dst = scores + (size_t)i * BSZ + j0 + hi * 8;
#pragma unroll
        for (int r = 0; r < 8; ++r) {
            dst[r]      = p0[r] + bb3;
            dst[16 + r] = p1[r] + bb3;
        }
    }
}

// ---------------------------------------------------------------------------
// Row-wise logsumexp over scores[i, :].
// ---------------------------------------------------------------------------
__global__ __launch_bounds__(256) void row_lse(
    const float* __restrict__ scores, float* __restrict__ lse)
{
    __shared__ float red[256];
    const int i = blockIdx.x;
    const float* row = scores + (size_t)i * BSZ;

    float m = -INFINITY;
    for (int j = threadIdx.x; j < BSZ; j += 256) m = fmaxf(m, row[j]);
    red[threadIdx.x] = m;
    __syncthreads();
    for (int s = 128; s > 0; s >>= 1) {
        if (threadIdx.x < s) red[threadIdx.x] = fmaxf(red[threadIdx.x], red[threadIdx.x + s]);
        __syncthreads();
    }
    const float mx = red[0];
    __syncthreads();

    float sum = 0.0f;
    for (int j = threadIdx.x; j < BSZ; j += 256) sum += __expf(row[j] - mx);
    red[threadIdx.x] = sum;
    __syncthreads();
    for (int s = 128; s > 0; s >>= 1) {
        if (threadIdx.x < s) red[threadIdx.x] += red[threadIdx.x + s];
        __syncthreads();
    }
    if (threadIdx.x == 0) lse[i] = mx + logf(red[0]);
}

// ---------------------------------------------------------------------------
// Final: out = -(log B + mean(diag) - mean(lse))
// ---------------------------------------------------------------------------
__global__ __launch_bounds__(512) void final_mi(
    const float* __restrict__ scores, const float* __restrict__ lse,
    float* __restrict__ out)
{
    __shared__ float rd[512], rl[512];
    const int i = threadIdx.x;
    rd[i] = scores[(size_t)i * BSZ + i];
    rl[i] = lse[i];
    __syncthreads();
    for (int s = 256; s > 0; s >>= 1) {
        if (i < s) { rd[i] += rd[i + s]; rl[i] += rl[i + s]; }
        __syncthreads();
    }
    if (i == 0) {
        const float mi = logf((float)BSZ) + rd[0] * (1.0f / BSZ) - rl[0] * (1.0f / BSZ);
        out[0] = -mi;
    }
}

extern "C" void kernel_launch(void* const* d_in, const int* in_sizes, int n_in,
                              void* d_out, int out_size, void* d_ws, size_t ws_size,
                              hipStream_t stream) {
    const float* X  = (const float*)d_in[0];
    const float* Y  = (const float*)d_in[1];
    const float* W1 = (const float*)d_in[2];
    const float* b1 = (const float*)d_in[3];
    const float* W2 = (const float*)d_in[4];
    const float* b2 = (const float*)d_in[5];
    const float* W3 = (const float*)d_in[6];
    const float* b3 = (const float*)d_in[7];
    float* out = (float*)d_out;

    char* ws = (char*)d_ws;
    float*  Ax     = (float*)(ws + 0);            // 512 KB
    float*  Ay     = (float*)(ws + 524288);       // 512 KB
    __bf16* W2p    = (__bf16*)(ws + 1048576);     // 128 KB
    float*  scores = (float*)(ws + 1179648);      // 1 MB
    float*  lse    = (float*)(ws + 2228224);      // 2 KB

    // Allow >64KB dynamic LDS for the main kernel (320 KB/WGP on CDNA5).
    hipFuncSetAttribute(reinterpret_cast<const void*>(mlp_scores),
                        hipFuncAttributeMaxDynamicSharedMemorySize, 131072);

    prep_axay<<<1024, 256, 0, stream>>>(X, Y, W1, b1, Ax, Ay);
    pack_w2<<<16, 256, 0, stream>>>(W2, W2p);
    mlp_scores<<<dim3(2, 512), 256, 131072, stream>>>(Ax, Ay, W2p, b2, W3, b3, scores);
    row_lse<<<512, 256, 0, stream>>>(scores, lse);
    final_mi<<<1, 512, 0, stream>>>(scores, lse, out);
}
